// lstm_62886911148478
// MI455X (gfx1250) — matrix operand order repaired and checked by reading, not compiled, not run on hardware
//
#include <hip/hip_runtime.h>

// ---------------------------------------------------------------------------
// 2-layer LSTM (B=256, T=2048, I=64, H=256, O=1) for gfx1250 (MI455X).
// Batch-partitioned persistent workgroups: 16 WGs x 16 waves (512 thr).
// Each WG owns a 16-row batch tile and runs the whole T loop locally.
// GEMMs via v_wmma_f32_16x16x32_bf16; weights pre-packed to bf16 B-fragments.
// Gate math fused in (f,i)/(g,o) pairs; x converted to bf16 cooperatively.
// ---------------------------------------------------------------------------

typedef __attribute__((ext_vector_type(16))) __bf16       v16bf;
typedef __attribute__((ext_vector_type(8)))  float        v8f;
typedef __attribute__((ext_vector_type(4)))  unsigned int v4u;
typedef __attribute__((ext_vector_type(2)))  float        v2f;

struct U8x32 { v4u a, b; };   // 32 bytes = one 16-element bf16 fragment

enum { TT = 2048, II = 64, HH = 256,
       HROW = 264,   // padded ushort row stride for h tiles
       XROW = 72 };  // padded ushort row stride for x tiles

static __device__ __forceinline__ unsigned short f2bf(float f) {
  unsigned u = __builtin_bit_cast(unsigned, f);
  u += 0x7fffu + ((u >> 16) & 1u);          // round-to-nearest-even
  return (unsigned short)(u >> 16);
}
static __device__ __forceinline__ unsigned pk2(float a, float b) {
  return (unsigned)f2bf(a) | ((unsigned)f2bf(b) << 16);
}
static __device__ __forceinline__ float bf2f(unsigned short u) {
  return __builtin_bit_cast(float, ((unsigned)u) << 16);
}
static __device__ __forceinline__ float sigm(float x) {
  return 1.0f / (1.0f + __expf(-x));
}
static __device__ __forceinline__ v8f splat8(float v) {
  v8f r;
#pragma unroll
  for (int e = 0; e < 8; ++e) r[e] = v;
  return r;
}

// B fragment: 32 contiguous bytes per lane (pre-packed; p already lane-offset)
static __device__ __forceinline__ v16bf ld_bfrag(const unsigned short* __restrict__ p) {
  U8x32 t;
  t.a = *(const v4u*)(p);
  t.b = *(const v4u*)(p + 8);
  return __builtin_bit_cast(v16bf, t);
}
// A fragment from row-major bf16 LDS tile: lane holds row (lane&15),
// K elements [c0, c0+8) and [c0+16, c0+24)  (ISA 16-bit A striping)
static __device__ __forceinline__ v16bf ld_afrag_lds(const unsigned short* p0) {
  U8x32 t;
  t.a = *(const v4u*)(p0);
  t.b = *(const v4u*)(p0 + 16);
  return __builtin_bit_cast(v16bf, t);
}

#define WMMA_BF16(acc, a, b) \
  (acc) = __builtin_amdgcn_wmma_f32_16x16x32_bf16(false, (a), false, (b), (short)0, (acc), false, false)

// Two gate tiles (nA, nB) of layer 1: x part (K=64, A cached in regs) then
// h1 part (K=256, A from LDS). pW pointers are pre-offset by lane*16.
static __device__ __forceinline__ void gemm_pair_l1(
    v8f& accA, v8f& accB, const v16bf ax[2],
    const unsigned short* __restrict__ pih1,
    const unsigned short* __restrict__ phh1,
    const unsigned short* hrow, int nA, int nB, int asel)
{
#pragma unroll
  for (int kb = 0; kb < 2; ++kb) {
    v16bf bA = ld_bfrag(pih1 + (nA * 2 + kb) * 512);
    v16bf bB = ld_bfrag(pih1 + (nB * 2 + kb) * 512);
    WMMA_BF16(accA, ax[kb], bA);
    WMMA_BF16(accB, ax[kb], bB);
  }
#pragma unroll 4
  for (int kb = 0; kb < 8; ++kb) {
    v16bf a  = ld_afrag_lds(hrow + kb * 32 + asel);
    v16bf bA = ld_bfrag(phh1 + (nA * 8 + kb) * 512);
    v16bf bB = ld_bfrag(phh1 + (nB * 8 + kb) * 512);
    WMMA_BF16(accA, a, bA);
    WMMA_BF16(accB, a, bB);
  }
}

// Two gate tiles of layer 2: h1_t part then h2_{t-1} part (both K=256, LDS).
static __device__ __forceinline__ void gemm_pair_l2(
    v8f& accA, v8f& accB,
    const unsigned short* __restrict__ pih2,
    const unsigned short* __restrict__ phh2,
    const unsigned short* h1row, const unsigned short* h2row,
    int nA, int nB, int asel)
{
#pragma unroll 4
  for (int kb = 0; kb < 8; ++kb) {
    v16bf a  = ld_afrag_lds(h1row + kb * 32 + asel);
    v16bf bA = ld_bfrag(pih2 + (nA * 8 + kb) * 512);
    v16bf bB = ld_bfrag(pih2 + (nB * 8 + kb) * 512);
    WMMA_BF16(accA, a, bA);
    WMMA_BF16(accB, a, bB);
  }
#pragma unroll 4
  for (int kb = 0; kb < 8; ++kb) {
    v16bf a  = ld_afrag_lds(h2row + kb * 32 + asel);
    v16bf bA = ld_bfrag(phh2 + (nA * 8 + kb) * 512);
    v16bf bB = ld_bfrag(phh2 + (nB * 8 + kb) * 512);
    WMMA_BF16(accA, a, bA);
    WMMA_BF16(accB, a, bB);
  }
}

// ---------------------------------------------------------------------------
// Prep: re-pack W[1024 x K] (row-major f32) into bf16 WMMA-B fragments.
//   value = W[ntile*16 + lane%16][kb*32 + (lane/16)*8 + (e%8) + (e/8)*16]
// ---------------------------------------------------------------------------
__global__ void pack_b_kernel(const float* __restrict__ W,
                              unsigned short* __restrict__ dst, int K) {
  int idx = blockIdx.x * blockDim.x + threadIdx.x;
  int total = 1024 * K;
  if (idx >= total) return;
  int e    = idx & 15;
  int lane = (idx >> 4) & 31;
  int frag = idx >> 9;
  int KB    = K >> 5;
  int ntile = frag / KB;
  int kb    = frag - ntile * KB;
  int n = ntile * 16 + (lane & 15);
  int k = kb * 32 + (lane >> 4) * 8 + (e & 7) + (e >> 3) * 16;
  dst[idx] = f2bf(W[(size_t)n * K + k]);
}

// ---------------------------------------------------------------------------
// Main persistent LSTM kernel. grid = 16 (batch tiles), block = 512 (16 waves).
// wave wv owns hidden tile wv -> gate tiles {i: wv, f: 16+wv, g: 32+wv, o: 48+wv}
// ---------------------------------------------------------------------------
__global__ __launch_bounds__(512, 1) void lstm_main(
    const float* __restrict__ x,
    const unsigned short* __restrict__ pih1g,
    const unsigned short* __restrict__ phh1g,
    const unsigned short* __restrict__ pih2g,
    const unsigned short* __restrict__ phh2g,
    const float* __restrict__ bih1, const float* __restrict__ bhh1,
    const float* __restrict__ bih2, const float* __restrict__ bhh2,
    const float* __restrict__ wlin, const float* __restrict__ blin,
    float* __restrict__ out)
{
  __shared__ __align__(16) unsigned short h1s[2][16][HROW];
  __shared__ __align__(16) unsigned short h2s[2][16][HROW];
  __shared__ __align__(16) unsigned short xs[2][16][XROW];
  __shared__ float wls[HH];

  const int tid  = threadIdx.x;
  const int lane = tid & 31;
  const int wv   = tid >> 5;          // hidden tile / wave id, 0..15
  const int wgb  = blockIdx.x * 16;   // batch row base for this WG

  for (int i = tid; i < 2 * 16 * HROW; i += 512) {
    ((unsigned short*)h1s)[i] = 0;
    ((unsigned short*)h2s)[i] = 0;
  }
  if (tid < HH) wls[tid] = wlin[tid];

  // cooperative x-tile conversion: thread -> (row = tid>>5, cols 2*(tid&31)+{0,1})
  const int xr = tid >> 5, xc = (tid & 31) * 2;
  const float* xcv = x + (size_t)(wgb + xr) * TT * II + xc;
  {
    v2f v = *(const v2f*)(xcv);                      // x at t = 0
    *(unsigned*)&xs[0][xr][xc] = pk2(v[0], v[1]);
  }
  __syncthreads();

  // per-lane gate biases (column-constant across the 16 batch rows)
  float b1[4], b2[4];
#pragma unroll
  for (int j = 0; j < 4; ++j) {
    int n = (j * 16 + wv) * 16 + (lane & 15);
    b1[j] = bih1[n] + bhh1[n];
    b2[j] = bih2[n] + bhh2[n];
  }
  const float bl = blin[0];

  float c1[8], c2[8];
#pragma unroll
  for (int e = 0; e < 8; ++e) { c1[e] = 0.f; c2[e] = 0.f; }

  const int arow = lane & 15;           // A-fragment row this lane serves
  const int asel = (lane >> 4) * 8;     // K sub-offset (ISA A striping)

  // pre-offset weight pointers by this lane's fragment slot
  const unsigned short* pih1 = pih1g + lane * 16;
  const unsigned short* phh1 = phh1g + lane * 16;
  const unsigned short* pih2 = pih2g + lane * 16;
  const unsigned short* phh2 = phh2g + lane * 16;

  const int nI = wv, nF = 16 + wv, nG = 32 + wv, nO = 48 + wv;

  for (int t = 0; t < TT; ++t) {
    const int cur = t & 1, nxt = cur ^ 1;

    // x_t A-fragments from LDS: load once, reuse across both layer-1 passes
    v16bf ax[2];
    ax[0] = ld_afrag_lds(&xs[cur][arow][asel]);
    ax[1] = ld_afrag_lds(&xs[cur][arow][32 + asel]);

    // ============================ layer 1 ============================
    {
      const unsigned short* hrow = &h1s[cur][arow][0];
      v8f accF = splat8(b1[1]), accI = splat8(b1[0]);
      gemm_pair_l1(accF, accI, ax, pih1, phh1, hrow, nF, nI, asel);
      float iv[8];
#pragma unroll
      for (int e = 0; e < 8; ++e) {
        c1[e] *= sigm(accF[e]);
        iv[e]  = sigm(accI[e]);
      }
      v8f accG = splat8(b1[2]), accO = splat8(b1[3]);
      gemm_pair_l1(accG, accO, ax, pih1, phh1, hrow, nG, nO, asel);
      unsigned short* hw = &h1s[nxt][0][wv * 16 + (lane & 15)];
#pragma unroll
      for (int e = 0; e < 8; ++e) {
        c1[e] += iv[e] * tanhf(accG[e]);
        float hv = sigm(accO[e]) * tanhf(c1[e]);
        int m = e + ((lane >> 4) << 3);       // C-layout row
        hw[m * HROW] = f2bf(hv);
      }
    }
    __syncthreads();

    // ============================ layer 2 ============================
    {
      const unsigned short* h1row = &h1s[nxt][arow][0];
      const unsigned short* h2row = &h2s[cur][arow][0];
      v8f accF = splat8(b2[1]), accI = splat8(b2[0]);
      gemm_pair_l2(accF, accI, pih2, phh2, h1row, h2row, nF, nI, asel);
      float iv[8];
#pragma unroll
      for (int e = 0; e < 8; ++e) {
        c2[e] *= sigm(accF[e]);
        iv[e]  = sigm(accI[e]);
      }
      v8f accG = splat8(b2[2]), accO = splat8(b2[3]);
      gemm_pair_l2(accG, accO, pih2, phh2, h1row, h2row, nG, nO, asel);
      unsigned short* hw = &h2s[nxt][0][wv * 16 + (lane & 15)];
#pragma unroll
      for (int e = 0; e < 8; ++e) {
        c2[e] += iv[e] * tanhf(accG[e]);
        float hv = sigm(accO[e]) * tanhf(c2[e]);
        int m = e + ((lane >> 4) << 3);
        hw[m * HROW] = f2bf(hv);
      }
    }

    // cooperative conversion of x(t+1) into the other x slot; fenced by the
    // barrier below before any step-(t+1) reads
    if (t + 1 < TT) {
      v2f v = *(const v2f*)(xcv + (size_t)(t + 1) * II);
      *(unsigned*)&xs[nxt][xr][xc] = pk2(v[0], v[1]);
    }
    __syncthreads();

    // ============================ output ============================
    // wave wv computes out[wgb+wv][t] = h2_t . W_lin + b_lin (wave32 reduce)
    {
      float s = 0.f;
      const unsigned short* hr = &h2s[nxt][wv][lane * 8];
#pragma unroll
      for (int e = 0; e < 8; ++e) s += bf2f(hr[e]) * wls[lane * 8 + e];
#pragma unroll
      for (int off = 16; off > 0; off >>= 1) s += __shfl_xor(s, off, 32);
      if (lane == 0) out[(size_t)(wgb + wv) * TT + t] = s + bl;
    }

    // prefetch x(t+2): 32 threads x 128B covers the 16x64 f32 tile exactly
    if (t + 2 < TT && tid < 32)
      __builtin_prefetch(x + ((size_t)(wgb + (tid >> 1)) * TT + (t + 2)) * II
                           + (tid & 1) * 32, 0, 3);
  }
}

// ---------------------------------------------------------------------------
extern "C" void kernel_launch(void* const* d_in, const int* in_sizes, int n_in,
                              void* d_out, int out_size, void* d_ws, size_t ws_size,
                              hipStream_t stream) {
  (void)in_sizes; (void)n_in; (void)out_size; (void)ws_size;
  const float* x    = (const float*)d_in[0];
  const float* Wih1 = (const float*)d_in[1];
  const float* Whh1 = (const float*)d_in[2];
  const float* bih1 = (const float*)d_in[3];
  const float* bhh1 = (const float*)d_in[4];
  const float* Wih2 = (const float*)d_in[5];
  const float* Whh2 = (const float*)d_in[6];
  const float* bih2 = (const float*)d_in[7];
  const float* bhh2 = (const float*)d_in[8];
  const float* Wlin = (const float*)d_in[9];
  const float* blin = (const float*)d_in[10];

  // workspace: packed bf16 weights, 1.66 MB total (stays resident in L2)
  unsigned short* p_ih1 = (unsigned short*)d_ws;            // 64*2*512
  unsigned short* p_hh1 = p_ih1 + 64 * 2 * 512;             // 64*8*512
  unsigned short* p_ih2 = p_hh1 + 64 * 8 * 512;
  unsigned short* p_hh2 = p_ih2 + 64 * 8 * 512;

  pack_b_kernel<<<(1024 * 64  + 255) / 256, 256, 0, stream>>>(Wih1, p_ih1, 64);
  pack_b_kernel<<<(1024 * 256 + 255) / 256, 256, 0, stream>>>(Whh1, p_hh1, 256);
  pack_b_kernel<<<(1024 * 256 + 255) / 256, 256, 0, stream>>>(Wih2, p_ih2, 256);
  pack_b_kernel<<<(1024 * 256 + 255) / 256, 256, 0, stream>>>(Whh2, p_hh2, 256);

  lstm_main<<<16, 512, 0, stream>>>(x, p_ih1, p_hh1, p_ih2, p_hh2,
                                    bih1, bhh1, bih2, bhh2, Wlin, blin,
                                    (float*)d_out);
}
